// GraphTransformerLayer_47330539602185
// MI455X (gfx1250) — matrix operand order repaired and checked by reading, not compiled, not run on hardware
//
#include <hip/hip_runtime.h>

typedef __attribute__((ext_vector_type(16))) __bf16 v16bf;
typedef __attribute__((ext_vector_type(8)))  float  v8f;
typedef __attribute__((ext_vector_type(4)))  int    v4i_t;

union FragAB { v16bf v; unsigned u[8]; };
union FragC  { v8f  v; float f[8]; };

// ---------------------------------------------------------------------------
// Async Global->LDS staging (CDNA5 GLOBAL_LOAD_ASYNC_TO_LDS_B128, ASYNCcnt),
// guarded so the kernel still compiles if this clang lacks the builtin.
// Builtin signature on this toolchain: (v4i addrspace(1)*, v4i addrspace(3)*,
// imm offset, imm cpol).
// ---------------------------------------------------------------------------
#if defined(__HIP_DEVICE_COMPILE__) && \
    __has_builtin(__builtin_amdgcn_global_load_async_to_lds_b128) && \
    __has_builtin(__builtin_amdgcn_s_wait_asynccnt)
#define USE_ASYNC_LDS 1
#else
#define USE_ASYNC_LDS 0
#endif

#if USE_ASYNC_LDS
typedef __attribute__((address_space(1))) v4i_t* g_v4i_p;
typedef __attribute__((address_space(3))) v4i_t* l_v4i_p;
#endif

__device__ __forceinline__ void copy16_g2lds(const __bf16* gsrc, __bf16* ldst) {
#if USE_ASYNC_LDS
    __builtin_amdgcn_global_load_async_to_lds_b128(
        (g_v4i_p)(gsrc), (l_v4i_p)(ldst), 0, 0);
#else
    *reinterpret_cast<uint4*>(ldst) = *reinterpret_cast<const uint4*>(gsrc);
#endif
}

__device__ __forceinline__ void wait_async_lds() {
#if USE_ASYNC_LDS
    __builtin_amdgcn_s_wait_asynccnt(0);
#endif
}

// ---------------------------------------------------------------------------
// fp32 -> bf16 convert (plain)
// ---------------------------------------------------------------------------
__global__ void cvt_f32_bf16_kernel(const float* __restrict__ x,
                                    __bf16* __restrict__ y, long long n) {
    long long i = (long long)blockIdx.x * blockDim.x + threadIdx.x;
    if (i < n) y[i] = (__bf16)x[i];
}

// fp32 [K,N] -> bf16 transposed [N,K] (weights, one-time)
__global__ void cvtT_f32_bf16_kernel(const float* __restrict__ x,
                                     __bf16* __restrict__ y, int K, int N) {
    int i = blockIdx.x * blockDim.x + threadIdx.x;
    if (i < K * N) {
        int k = i / N, n = i - k * N;
        y[(long long)n * K + k] = (__bf16)x[i];
    }
}

// ---------------------------------------------------------------------------
// Tiled bf16 WMMA GEMM: out[M,N] = A[M,K] @ B[K,N] (+bias)(+relu)(+res)
//   A  : bf16 row-major [M,K]
//   BT : bf16 TRANSPOSED weights [N,K] (so both tiles are contiguous b128s)
// Block tile 128x64, 256 threads = 8 waves, each wave 32x32 (2x2 fragments).
// Double-buffered LDS; staging via async Global->LDS when available.
// ---------------------------------------------------------------------------
#define BM    128
#define BNT   64
#define BK    32
#define LDA_S 40   // row stride (elems): 80B, 16B-aligned, bank-spread
#define LDB_S 40

template<bool RELU, bool BIAS, bool RES, bool OUTBF>
__global__ __launch_bounds__(256) void gemm_bf16_wmma(
    const __bf16* __restrict__ A, const __bf16* __restrict__ BT,
    const float* __restrict__ bias, const float* __restrict__ res,
    void* __restrict__ outv, int M, int N, int K)
{
    __shared__ __align__(16) __bf16 As[2][BM * LDA_S];
    __shared__ __align__(16) __bf16 Bs[2][BNT * LDB_S];

    const int tid  = threadIdx.x;
    const int lane = tid & 31;
    const int wv   = tid >> 5;
    const int wm   = wv & 3;      // 4 waves along M
    const int wn   = wv >> 2;     // 2 waves along N
    const int m0   = blockIdx.x * BM;
    const int n0   = blockIdx.y * BNT;
    const int l16  = lane & 15;
    const int half = (lane >> 4);

    // A tile: 128 rows x 32 elems = 512 chunks of 8 bf16 (16B)
    auto stageA = [&](int buf, int k0) {
        for (int g = tid; g < (BM * BK) / 8; g += 256) {
            int row  = g >> 2;
            int ck   = (g & 3) * 8;
            int grow = m0 + row;
            __bf16* dst = &As[buf][row * LDA_S + ck];
            if (grow < M) {
                copy16_g2lds(A + (long long)grow * K + k0 + ck, dst);
            } else {
                uint4 z; z.x = z.y = z.z = z.w = 0u;
                *reinterpret_cast<uint4*>(dst) = z;
            }
        }
    };
    // B tile (from transposed weights): 64 cols x 32 elems = 256 chunks
    auto stageB = [&](int buf, int k0) {
        for (int g = tid; g < (BNT * BK) / 8; g += 256) {
            int col = g >> 2;
            int ck  = (g & 3) * 8;
            copy16_g2lds(BT + (long long)(n0 + col) * K + k0 + ck,
                         &Bs[buf][col * LDB_S + ck]);
        }
    };

    FragC c00, c01, c10, c11;
#pragma unroll
    for (int i = 0; i < 8; ++i) { c00.f[i]=0.f; c01.f[i]=0.f; c10.f[i]=0.f; c11.f[i]=0.f; }

    stageA(0, 0);
    stageB(0, 0);
    wait_async_lds();
    __syncthreads();

    const int nk = K / BK;
    for (int kt = 0; kt < nk; ++kt) {
        const int cur = kt & 1;
        if (kt + 1 < nk) {            // prefetch next tile into the other buffer
            stageA(cur ^ 1, (kt + 1) * BK);
            stageB(cur ^ 1, (kt + 1) * BK);
        }

        // ---- fragments: contiguous k = 16*half + s for both A and B ----
        FragAB a0, a1, b0, b1;
        {
            const int kb = half ? 16 : 0;
            const int r0 = (wm * 32 + l16) * LDA_S;
            const int r1 = (wm * 32 + 16 + l16) * LDA_S;
#pragma unroll
            for (int i = 0; i < 4; ++i) {
                a0.u[i]     = *reinterpret_cast<const unsigned*>(&As[cur][r0 + kb + 2*i]);
                a0.u[4 + i] = *reinterpret_cast<const unsigned*>(&As[cur][r0 + kb + 8 + 2*i]);
                a1.u[i]     = *reinterpret_cast<const unsigned*>(&As[cur][r1 + kb + 2*i]);
                a1.u[4 + i] = *reinterpret_cast<const unsigned*>(&As[cur][r1 + kb + 8 + 2*i]);
            }
            const int kbb = half ? 16 : 0;
            const int q0  = (wn * 32 + l16) * LDB_S;
            const int q1  = (wn * 32 + 16 + l16) * LDB_S;
#pragma unroll
            for (int i = 0; i < 8; ++i) {
                b0.u[i] = *reinterpret_cast<const unsigned*>(&Bs[cur][q0 + kbb + 2*i]);
                b1.u[i] = *reinterpret_cast<const unsigned*>(&Bs[cur][q1 + kbb + 2*i]);
            }
        }
        c00.v = __builtin_amdgcn_wmma_f32_16x16x32_bf16(false, a0.v, false, b0.v, (short)0, c00.v, false, false);
        c01.v = __builtin_amdgcn_wmma_f32_16x16x32_bf16(false, a0.v, false, b1.v, (short)0, c01.v, false, false);
        c10.v = __builtin_amdgcn_wmma_f32_16x16x32_bf16(false, a1.v, false, b0.v, (short)0, c10.v, false, false);
        c11.v = __builtin_amdgcn_wmma_f32_16x16x32_bf16(false, a1.v, false, b1.v, (short)0, c11.v, false, false);

        wait_async_lds();     // next tile's async DMA done before we flip buffers
        __syncthreads();
    }

    // ---- epilogue: C layout = vgpr r -> row r + 8*(lane>=16), col = lane%16 ----
    const int rl = half * 8;
    const int cl = l16;
#pragma unroll
    for (int fi = 0; fi < 2; ++fi) {
#pragma unroll
        for (int fj = 0; fj < 2; ++fj) {
            FragC& c = fi ? (fj ? c11 : c10) : (fj ? c01 : c00);
            const int rowb = m0 + wm * 32 + fi * 16 + rl;
            const int col  = n0 + wn * 32 + fj * 16 + cl;
#pragma unroll
            for (int r = 0; r < 8; ++r) {
                int row = rowb + r;
                if (row < M) {
                    float v = c.f[r];
                    if (BIAS) v += bias[col];
                    if (RELU) v = fmaxf(v, 0.f);
                    long long idx = (long long)row * N + col;
                    if (RES)  v += res[idx];
                    if (OUTBF) ((__bf16*)outv)[idx] = (__bf16)v;
                    else       ((float*)outv)[idx]  = v;
                }
            }
        }
    }
}

// ---------------------------------------------------------------------------
// Edge attention: score = K[src]*Q[dst]*pe/sqrt(16); w = exp(clip(sum_head));
// atomic segment-sum of w*V[src] and w into dst. Q/K/V are L2-resident.
// ---------------------------------------------------------------------------
__global__ void edge_attn_kernel(const float* __restrict__ Q,
                                 const float* __restrict__ Kf,
                                 const float* __restrict__ V,
                                 const __bf16* __restrict__ pe,
                                 const int* __restrict__ src,
                                 const int* __restrict__ dst,
                                 __bf16* __restrict__ e_attn,
                                 float* __restrict__ wV,
                                 float* __restrict__ z,
                                 long long total)
{
    const long long stride = (long long)gridDim.x * blockDim.x;
    for (long long t = (long long)blockIdx.x * blockDim.x + threadIdx.x;
         t < total; t += stride) {
        const int c = (int)(t & 127);
        const long long eid = t >> 7;
        const int s = src[eid];
        const int d = dst[eid];
        const float sc = Kf[(long long)s * 128 + c] * Q[(long long)d * 128 + c]
                         * 0.25f * (float)pe[t];
        e_attn[t] = (__bf16)sc;
        float sum = sc;
#pragma unroll
        for (int o = 8; o; o >>= 1) sum += __shfl_xor(sum, o, 16);
        const float w = __expf(fminf(fmaxf(sum, -5.f), 5.f));
        atomicAdd(&wV[(long long)d * 128 + c], V[(long long)s * 128 + c] * w);
        if ((c & 15) == 0) atomicAdd(&z[(long long)d * 8 + (c >> 4)], w);
    }
}

// h_attn = wV / (z + 1e-6), store bf16 for next GEMM
__global__ void attn_norm_kernel(const float* __restrict__ wV,
                                 const float* __restrict__ z,
                                 __bf16* __restrict__ out, long long total)
{
    const long long stride = (long long)gridDim.x * blockDim.x;
    for (long long i = (long long)blockIdx.x * blockDim.x + threadIdx.x;
         i < total; i += stride) {
        const int c = (int)(i & 127);
        const long long n = i >> 7;
        out[i] = (__bf16)(wV[i] / (z[n * 8 + (c >> 4)] + 1e-6f));
    }
}

// ---------------------------------------------------------------------------
// BatchNorm (training-mode, biased variance) over rows, 128 columns
// ---------------------------------------------------------------------------
__global__ void bn_stats_kernel(const float* __restrict__ x, int M,
                                float* __restrict__ sums, float* __restrict__ sqs)
{
    const int c = threadIdx.x;   // 128 threads
    float s = 0.f, q = 0.f;
    for (int r = blockIdx.x; r < M; r += gridDim.x) {
        float v = x[(long long)r * 128 + c];
        s += v; q += v * v;
    }
    atomicAdd(&sums[c], s);
    atomicAdd(&sqs[c], q);
}

template<bool WBF>
__global__ void bn_apply_kernel(const float* __restrict__ x, float* __restrict__ y,
                                __bf16* __restrict__ ybf, int M,
                                const float* __restrict__ sums,
                                const float* __restrict__ sqs,
                                const float* __restrict__ g,
                                const float* __restrict__ b)
{
    const long long total  = (long long)M * 128;
    const long long stride = (long long)gridDim.x * blockDim.x;
    const float invM = 1.f / (float)M;
    for (long long i = (long long)blockIdx.x * blockDim.x + threadIdx.x;
         i < total; i += stride) {
        const int c = (int)(i & 127);
        const float m   = sums[c] * invM;
        const float var = sqs[c] * invM - m * m;
        const float o = (x[i] - m) * rsqrtf(var + 1e-5f) * g[c] + b[c];
        y[i] = o;
        if (WBF) ybf[i] = (__bf16)o;
    }
}

// ---------------------------------------------------------------------------
// Host orchestration
// ---------------------------------------------------------------------------
extern "C" void kernel_launch(void* const* d_in, const int* in_sizes, int n_in,
                              void* d_out, int out_size, void* d_ws, size_t ws_size,
                              hipStream_t stream)
{
    (void)in_sizes; (void)n_in; (void)out_size; (void)ws_size;
    const int N = 50000, E = 640000, D = 128, DFF = 256;

    const float* h    = (const float*)d_in[0];
    const float* e    = (const float*)d_in[1];
    const int*   src  = (const int*)d_in[2];
    const int*   dst  = (const int*)d_in[3];
    const float* Wq   = (const float*)d_in[4];
    const float* Wk   = (const float*)d_in[5];
    const float* Wv   = (const float*)d_in[6];
    const float* We   = (const float*)d_in[7];
    const float* Wo_h = (const float*)d_in[8];
    const float* bo_h = (const float*)d_in[9];
    const float* Wo_e = (const float*)d_in[10];
    const float* bo_e = (const float*)d_in[11];
    const float* g1h  = (const float*)d_in[12];
    const float* b1h  = (const float*)d_in[13];
    const float* g1e  = (const float*)d_in[14];
    const float* b1e  = (const float*)d_in[15];
    const float* Wf1h = (const float*)d_in[16];
    const float* bf1h = (const float*)d_in[17];
    const float* Wf2h = (const float*)d_in[18];
    const float* bf2h = (const float*)d_in[19];
    const float* Wf1e = (const float*)d_in[20];
    const float* bf1e = (const float*)d_in[21];
    const float* Wf2e = (const float*)d_in[22];
    const float* bf2e = (const float*)d_in[23];
    const float* g2h  = (const float*)d_in[24];
    const float* b2h  = (const float*)d_in[25];
    const float* g2e  = (const float*)d_in[26];
    const float* b2e  = (const float*)d_in[27];

    // workspace carve
    size_t off = 0;
    auto carve = [&](size_t bytes) -> void* {
        off = (off + 255) & ~(size_t)255;
        void* p = (char*)d_ws + off;
        off += bytes;
        return p;
    };
    // transposed bf16 weights [N,K]
    __bf16* WqT   = (__bf16*)carve((size_t)D * D * 2);
    __bf16* WkT   = (__bf16*)carve((size_t)D * D * 2);
    __bf16* WvT   = (__bf16*)carve((size_t)D * D * 2);
    __bf16* WeT   = (__bf16*)carve((size_t)D * D * 2);
    __bf16* WohT  = (__bf16*)carve((size_t)D * D * 2);
    __bf16* WoeT  = (__bf16*)carve((size_t)D * D * 2);
    __bf16* Wf1hT = (__bf16*)carve((size_t)D * DFF * 2);
    __bf16* Wf2hT = (__bf16*)carve((size_t)DFF * D * 2);
    __bf16* Wf1eT = (__bf16*)carve((size_t)D * DFF * 2);
    __bf16* Wf2eT = (__bf16*)carve((size_t)DFF * D * 2);
    // bf16 activations
    __bf16* h_bf    = (__bf16*)carve((size_t)N * D * 2);
    __bf16* e_bf    = (__bf16*)carve((size_t)E * D * 2);
    __bf16* pe_bf   = (__bf16*)carve((size_t)E * D * 2);
    __bf16* eat_bf  = (__bf16*)carve((size_t)E * D * 2);
    __bf16* hat_bf  = (__bf16*)carve((size_t)N * D * 2);
    __bf16* hhbn_bf = (__bf16*)carve((size_t)N * D * 2);
    __bf16* eebn_bf = (__bf16*)carve((size_t)E * D * 2);
    __bf16* t_h     = (__bf16*)carve((size_t)N * DFF * 2);
    __bf16* t_e     = (__bf16*)carve((size_t)E * DFF * 2);
    // fp32 buffers
    float* Qf   = (float*)carve((size_t)N * D * 4);
    float* Kf   = (float*)carve((size_t)N * D * 4);
    float* Vf   = (float*)carve((size_t)N * D * 4);
    float* wV   = (float*)carve((size_t)N * D * 4);
    float* zacc = (float*)carve((size_t)N * 8 * 4);
    float* hh   = (float*)carve((size_t)N * D * 4);
    float* ee   = (float*)carve((size_t)E * D * 4);
    float* stats = (float*)carve(256 * 4);

    float* out_h = (float*)d_out;
    float* out_e = out_h + (size_t)N * D;

    // zero atomic accumulators
    (void)hipMemsetAsync(wV,   0, (size_t)N * D * 4, stream);
    (void)hipMemsetAsync(zacc, 0, (size_t)N * 8 * 4, stream);

    // one-time weight transpose+convert, activation converts
    auto cvtT = [&](const float* s, __bf16* dptr, int K_, int N_) {
        int n = K_ * N_;
        cvtT_f32_bf16_kernel<<<(n + 255) / 256, 256, 0, stream>>>(s, dptr, K_, N_);
    };
    cvtT(Wq, WqT, D, D);       cvtT(Wk, WkT, D, D);
    cvtT(Wv, WvT, D, D);       cvtT(We, WeT, D, D);
    cvtT(Wo_h, WohT, D, D);    cvtT(Wo_e, WoeT, D, D);
    cvtT(Wf1h, Wf1hT, D, DFF); cvtT(Wf2h, Wf2hT, DFF, D);
    cvtT(Wf1e, Wf1eT, D, DFF); cvtT(Wf2e, Wf2eT, DFF, D);
    {
        long long nh = (long long)N * D, ne = (long long)E * D;
        cvt_f32_bf16_kernel<<<(unsigned)((nh + 255) / 256), 256, 0, stream>>>(h, h_bf, nh);
        cvt_f32_bf16_kernel<<<(unsigned)((ne + 255) / 256), 256, 0, stream>>>(e, e_bf, ne);
    }

    const dim3 blk(256);
    const dim3 gN128((N + BM - 1) / BM, D / BNT);     // (391, 2)
    const dim3 gN256((N + BM - 1) / BM, DFF / BNT);   // (391, 4)
    const dim3 gE128((E + BM - 1) / BM, D / BNT);     // (5000, 2)
    const dim3 gE256((E + BM - 1) / BM, DFF / BNT);   // (5000, 4)

    // Q = h@Wq, K = h@Wk, V = h@Wv  (fp32 out, fits L2 for edge gathers)
    gemm_bf16_wmma<false,false,false,false><<<gN128, blk, 0, stream>>>(h_bf, WqT, nullptr, nullptr, Qf, N, D, D);
    gemm_bf16_wmma<false,false,false,false><<<gN128, blk, 0, stream>>>(h_bf, WkT, nullptr, nullptr, Kf, N, D, D);
    gemm_bf16_wmma<false,false,false,false><<<gN128, blk, 0, stream>>>(h_bf, WvT, nullptr, nullptr, Vf, N, D, D);
    // pe = e@We  (bf16 out)
    gemm_bf16_wmma<false,false,false,true><<<gE128, blk, 0, stream>>>(e_bf, WeT, nullptr, nullptr, pe_bf, E, D, D);

    // edge scores + segment sums
    edge_attn_kernel<<<320000, 256, 0, stream>>>(Qf, Kf, Vf, pe_bf, src, dst,
                                                 eat_bf, wV, zacc, (long long)E * D);
    // h_attn = wV/(z+1e-6)
    attn_norm_kernel<<<25000, 256, 0, stream>>>(wV, zacc, hat_bf, (long long)N * D);

    // hh = h_attn@Wo_h + bo_h + h ; ee = e_attn@Wo_e + bo_e + e
    gemm_bf16_wmma<false,true,true,false><<<gN128, blk, 0, stream>>>(hat_bf, WohT, bo_h, h, hh, N, D, D);
    gemm_bf16_wmma<false,true,true,false><<<gE128, blk, 0, stream>>>(eat_bf, WoeT, bo_e, e, ee, E, D, D);

    auto run_bn = [&](float* x, __bf16* xbf, int M, const float* g, const float* b) {
        (void)hipMemsetAsync(stats, 0, 256 * 4, stream);
        bn_stats_kernel<<<1024, 128, 0, stream>>>(x, M, stats, stats + 128);
        if (xbf) bn_apply_kernel<true><<<4096, 256, 0, stream>>>(x, x, xbf, M, stats, stats + 128, g, b);
        else     bn_apply_kernel<false><<<4096, 256, 0, stream>>>(x, x, nullptr, M, stats, stats + 128, g, b);
    };
    run_bn(hh, hhbn_bf, N, g1h, b1h);   // hh -> hh_bn (in place) + bf16 copy
    run_bn(ee, eebn_bf, E, g1e, b1e);   // ee -> ee_bn (in place) + bf16 copy

    // FFN node: t = relu(hh_bn@Wf1h + bf1h) ; out_h = t@Wf2h + bf2h + hh_bn
    gemm_bf16_wmma<true,true,false,true><<<gN256, blk, 0, stream>>>(hhbn_bf, Wf1hT, bf1h, nullptr, t_h, N, DFF, D);
    gemm_bf16_wmma<false,true,true,false><<<gN128, blk, 0, stream>>>(t_h, Wf2hT, bf2h, hh, out_h, N, D, DFF);
    // FFN edge
    gemm_bf16_wmma<true,true,false,true><<<gE256, blk, 0, stream>>>(eebn_bf, Wf1eT, bf1e, nullptr, t_e, E, DFF, D);
    gemm_bf16_wmma<false,true,true,false><<<gE128, blk, 0, stream>>>(t_e, Wf2eT, bf2e, ee, out_e, E, D, DFF);

    // final BN directly on d_out segments
    run_bn(out_h, nullptr, N, g2h, b2h);
    run_bn(out_e, nullptr, E, g2e, b2e);
}
